// MultiHeadHigherOrderAttention_19696720019803
// MI455X (gfx1250) — matrix operand order repaired and checked by reading, not compiled, not run on hardware
//
#include <hip/hip_runtime.h>

// ---------------------------------------------------------------------------
// MI455X (gfx1250) implementation of bipartite multi-head GAT attention.
//
//   proj_kernel       : hs = in1 @ Ws, ht = in2 @ Wt (per head), f16 copies,
//                       and additive scores ss[h,s], st[h,t]
//   row_stats_kernel  : online softmax (max, sumexp) over s per (h, t)
//   col_stats_kernel  : online softmax (max, sumexp) over t per (h, s)
//   attn_wmma_kernel  : flash-style masked-softmax matmul using
//                       v_wmma_f32_16x16x32_f16; probabilities synthesized
//                       in-register; mask tiles streamed into LDS with
//                       double-buffered global_load_async_to_lds (ASYNCcnt).
// ---------------------------------------------------------------------------

typedef __attribute__((ext_vector_type(16))) _Float16 v16h;
typedef __attribute__((ext_vector_type(8)))  float    v8f;

#define N_S   8192
#define N_T   4096
#define NH    4
#define DOUT  64
#define DIN   128
#define ALPHA 0.1f

__device__ __forceinline__ float lrelu(float x) { return x > 0.f ? x : ALPHA * x; }

// ---------------------------------------------------------------------------
// Kernel 1: projections + scores.  grid = (N_S, NH, 2), block = 64.
// ---------------------------------------------------------------------------
__global__ __launch_bounds__(64)
void proj_kernel(const float* __restrict__ in1, const float* __restrict__ in2,
                 const float* __restrict__ Ws,  const float* __restrict__ Wt,
                 const float* __restrict__ a_src, const float* __restrict__ a_tgt,
                 _Float16* __restrict__ hs_h, _Float16* __restrict__ ht_h,
                 float* __restrict__ ss, float* __restrict__ st)
{
    const int z     = blockIdx.z;
    const int nRows = z ? N_T : N_S;
    const int row   = blockIdx.x;
    if (row >= nRows) return;
    const int h = blockIdx.y;

    const float* X  = z ? in2 : in1;
    const float* W  = (z ? Wt : Ws) + (size_t)h * DIN * DOUT;
    const float* av = (z ? a_tgt : a_src) + h * DOUT;
    _Float16*    Ho = (z ? ht_h : hs_h) + ((size_t)h * nRows + row) * DOUT;
    float*       sc = (z ? st : ss) + (size_t)h * nRows;

    __shared__ float xrow[DIN];
    __shared__ float red[64];

    const int o = threadIdx.x;
    xrow[o]      = X[(size_t)row * DIN + o];
    xrow[o + 64] = X[(size_t)row * DIN + o + 64];
    __syncthreads();

    float acc = 0.f;
#pragma unroll 8
    for (int i = 0; i < DIN; ++i)
        acc = fmaf(xrow[i], W[(size_t)i * DOUT + o], acc);

    Ho[o]  = (_Float16)acc;
    red[o] = acc * av[o];
    __syncthreads();
    for (int off = 32; off > 0; off >>= 1) {
        if (o < off) red[o] += red[o + off];
        __syncthreads();
    }
    if (o == 0) sc[row] = red[0];
}

// ---------------------------------------------------------------------------
// Kernel 2: per-target-row online softmax stats.  grid = N_T, block = 256.
// ---------------------------------------------------------------------------
__global__ __launch_bounds__(256)
void row_stats_kernel(const float* __restrict__ A,
                      const float* __restrict__ st, const float* __restrict__ ss,
                      float* __restrict__ m_ts, float* __restrict__ d_ts)
{
    const int t   = blockIdx.x;
    const int tid = threadIdx.x;

    float stv[NH];
#pragma unroll
    for (int h = 0; h < NH; ++h) stv[h] = st[(size_t)h * N_T + t];

    float m[NH], d[NH];
#pragma unroll
    for (int h = 0; h < NH; ++h) { m[h] = -1e30f; d[h] = 0.f; }

    for (int s = tid; s < N_S; s += 256) {
        float a = A[(size_t)t * N_S + s];
        if (a != 0.f) {
#pragma unroll
            for (int h = 0; h < NH; ++h) {
                float e = lrelu(stv[h] + ss[(size_t)h * N_S + s]);
                if (e > m[h]) { d[h] = d[h] * __expf(m[h] - e) + 1.f; m[h] = e; }
                else          { d[h] += __expf(e - m[h]); }
            }
        }
    }

    __shared__ float sm[256], sd[256];
    for (int h = 0; h < NH; ++h) {
        sm[tid] = m[h]; sd[tid] = d[h];
        __syncthreads();
        for (int off = 128; off > 0; off >>= 1) {
            if (tid < off) {
                float ma = sm[tid], mb = sm[tid + off];
                float da = sd[tid], db = sd[tid + off];
                float M = fmaxf(ma, mb);
                float D = 0.f;
                if (ma > -1e29f) D += da * __expf(ma - M);
                if (mb > -1e29f) D += db * __expf(mb - M);
                sm[tid] = M; sd[tid] = D;
            }
            __syncthreads();
        }
        if (tid == 0) {
            m_ts[(size_t)h * N_T + t] = sm[0];
            d_ts[(size_t)h * N_T + t] = sd[0];
        }
        __syncthreads();
    }
}

// ---------------------------------------------------------------------------
// Kernel 3: per-source-column online softmax stats.  grid = N_S/256, block=256.
// ---------------------------------------------------------------------------
__global__ __launch_bounds__(256)
void col_stats_kernel(const float* __restrict__ A,
                      const float* __restrict__ st, const float* __restrict__ ss,
                      float* __restrict__ m_st, float* __restrict__ d_st)
{
    const int s = blockIdx.x * 256 + threadIdx.x;

    float ssv[NH];
#pragma unroll
    for (int h = 0; h < NH; ++h) ssv[h] = ss[(size_t)h * N_S + s];

    float m[NH], d[NH];
#pragma unroll
    for (int h = 0; h < NH; ++h) { m[h] = -1e30f; d[h] = 0.f; }

    for (int t = 0; t < N_T; ++t) {
        float a = A[(size_t)t * N_S + s];
        if (a != 0.f) {
#pragma unroll
            for (int h = 0; h < NH; ++h) {
                float e = lrelu(st[(size_t)h * N_T + t] + ssv[h]);
                if (e > m[h]) { d[h] = d[h] * __expf(m[h] - e) + 1.f; m[h] = e; }
                else          { d[h] += __expf(e - m[h]); }
            }
        }
    }
#pragma unroll
    for (int h = 0; h < NH; ++h) {
        m_st[(size_t)h * N_S + s] = m[h];
        d_st[(size_t)h * N_S + s] = d[h];
    }
}

// ---------------------------------------------------------------------------
// Async mask-tile stager: copies a 64(M) x 32(K) f32 tile of the mask into
// LDS at ldsBase (layout aT[row*32+k]) using GLOBAL_LOAD_ASYNC_TO_LDS.
// !transA: contiguous rows -> 4 x b128 per wave.  transA: per-lane scatter
// (coalesced in global along M) -> 16 x b32 per wave.
// ---------------------------------------------------------------------------
__device__ __forceinline__ void stage_mask_async(const float* __restrict__ A,
                                                 int m0, int k0, int transA,
                                                 int tid, unsigned ldsBase)
{
    if (!transA) {
#pragma unroll
        for (int c = 0; c < 4; ++c) {
            int idx = tid + c * 128;            // 512 float4 chunks
            int row = idx >> 3, kq = (idx & 7) << 2;
            unsigned long long g =
                (unsigned long long)(A + (size_t)(m0 + row) * N_S + (k0 + kq));
            unsigned lds = ldsBase + (unsigned)((((row << 5) + kq)) << 2);
            asm volatile("global_load_async_to_lds_b128 %0, %1, off"
                         :: "v"(lds), "v"(g) : "memory");
        }
    } else {
#pragma unroll
        for (int c = 0; c < 16; ++c) {
            int idx = tid + c * 128;            // 2048 floats
            int k = idx >> 6, row = idx & 63;   // coalesced over row
            unsigned long long g =
                (unsigned long long)(A + (size_t)(k0 + k) * N_S + (m0 + row));
            unsigned lds = ldsBase + (unsigned)((((row << 5) + k)) << 2);
            asm volatile("global_load_async_to_lds_b32 %0, %1, off"
                         :: "v"(lds), "v"(g) : "memory");
        }
    }
}

// ---------------------------------------------------------------------------
// Kernel 4/5: out[m, h*64+n] = elu( sum_k P[m,k] * B[h,k,n] ),
//   P[m,k] = mask ? exp(lrelu(rowS[m]+colS[k]) - mArr[m]) / dArr[m] : 0
// transA==0 : M = t rows, K = s.   transA==1 : M = s rows, K = t.
// grid = (Mtot/64, NH), block = 128 (4 waves; wave w owns M-subtile 16*w).
// Mask tiles double-buffered in LDS via async loads (ASYNCcnt).
// ---------------------------------------------------------------------------
__global__ __launch_bounds__(128)
void attn_wmma_kernel(const float* __restrict__ A, int transA,
                      const float* __restrict__ rowS, const float* __restrict__ colS,
                      const float* __restrict__ mArr, const float* __restrict__ dArr,
                      const _Float16* __restrict__ B, int Mtot, int Ktot,
                      float* __restrict__ out, int outStride)
{
    const int h   = blockIdx.y;
    const int m0  = blockIdx.x * 64;
    const int tid = threadIdx.x;
    const int w   = tid >> 5;     // wave id 0..3
    const int l   = tid & 31;     // lane   0..31

    __shared__ float    aT[2][64 * 32];  // double-buffered mask tile [row*32+k]
    __shared__ _Float16 bT[DOUT * 32];   // bT[n*32 + k]  (B tile, transposed)
    __shared__ float    sRow[64], sM[64], sDinv[64];
    __shared__ float    sCol[32];

    if (tid < 64) {
        const size_t gi = (size_t)h * Mtot + m0 + tid;
        sRow[tid] = rowS[gi];
        sM[tid]   = mArr[gi];
        float dv  = dArr[gi];
        sDinv[tid] = dv > 0.f ? 1.f / dv : 0.f;
    }

    const unsigned aBase0 = (unsigned)(unsigned long long)(&aT[0][0]);
    const unsigned aBase1 = (unsigned)(unsigned long long)(&aT[1][0]);

    const _Float16* Bh = B + (size_t)h * Ktot * DOUT;

    // B staging decomposition: thread covers rows k=(tid>>3)+{0,16}, cols n0..n0+7
    const int bK = tid >> 3;            // 0..15
    const int bN = (tid & 7) << 3;      // 0,8,..,56

    v8f acc0 = {}, acc1 = {}, acc2 = {}, acc3 = {};

    // Prologue: kick off tile 0 into buffer 0.
    stage_mask_async(A, m0, 0, transA, tid, aBase0);

    for (int k0 = 0; k0 < Ktot; k0 += 32) {
        const int buf = (k0 >> 5) & 1;
        __syncthreads();   // previous iteration's LDS readers done

        if (tid < 32) sCol[tid] = colS[(size_t)h * Ktot + k0 + tid];

        // Stage B tile transposed: bT[n*32+k] = B[h, k0+k, n].
        // Vectorized global side: one uint4 (8 halves, 16B-aligned) per chunk,
        // then 8 unrolled ds_store_b16 scatters.
#pragma unroll
        for (int c = 0; c < 2; ++c) {
            const int k = bK + c * 16;
            const uint4 vv = *(const uint4*)(Bh + (size_t)(k0 + k) * DOUT + bN);
            const _Float16* hp = (const _Float16*)&vv;
#pragma unroll
            for (int j = 0; j < 8; ++j)
                bT[(bN + j) * 32 + k] = hp[j];
        }

        // Kick off next mask tile into the other buffer, then wait for the
        // current tile's async ops (in-order completion on ASYNCcnt).
        const int hasNext = (k0 + 32 < Ktot);
        if (hasNext)
            stage_mask_async(A, m0, k0 + 32, transA, tid, buf ? aBase0 : aBase1);
        if (!transA) {
            if (hasNext) asm volatile("s_wait_asynccnt 4"  ::: "memory");
            else         asm volatile("s_wait_asynccnt 0"  ::: "memory");
        } else {
            if (hasNext) asm volatile("s_wait_asynccnt 16" ::: "memory");
            else         asm volatile("s_wait_asynccnt 0"  ::: "memory");
        }
        __syncthreads();   // async LDS writes + bT visible to all waves

        const float* aBuf = &aT[buf][0];

        // --- Build probability A-fragment (16x32 f16 layout) -------------
        // lanes 0..15 : M = l,     K = {0..7, 16..23}
        // lanes 16..31: M = l-16,  K = {8..15, 24..31}
        const int   mloc = (l & 15) + 16 * w;
        const int   kbA  = (l < 16) ? 0 : 8;
        const float rs = sRow[mloc], mm = sM[mloc], di = sDinv[mloc];
        v16h pf;
#pragma unroll
        for (int j = 0; j < 16; ++j) {
            int v = j >> 1;
            int k = ((v < 4) ? 0 : 16) + kbA + 2 * (v & 3) + (j & 1);
            float a = aBuf[mloc * 32 + k];
            float p = 0.f;
            if (a != 0.f) {
                float e = lrelu(rs + sCol[k]);
                p = __expf(e - mm) * di;
            }
            pf[j] = (_Float16)p;
        }

        // --- Preload all four B fragments, then back-to-back WMMAs -------
        const int kbB = (l < 16) ? 0 : 8;
        const int nb  = (l & 15);
        v16h bf0, bf1, bf2, bf3;
#pragma unroll
        for (int j = 0; j < 16; ++j) {
            bf0[j] = bT[(nb +  0) * 32 + kbB + (j & 7) + ((j >> 3) << 4)];
            bf1[j] = bT[(nb + 16) * 32 + kbB + (j & 7) + ((j >> 3) << 4)];
            bf2[j] = bT[(nb + 32) * 32 + kbB + (j & 7) + ((j >> 3) << 4)];
            bf3[j] = bT[(nb + 48) * 32 + kbB + (j & 7) + ((j >> 3) << 4)];
        }
        acc0 = __builtin_amdgcn_wmma_f32_16x16x32_f16(false, pf, false, bf0, (short)0, acc0, false, false);
        acc1 = __builtin_amdgcn_wmma_f32_16x16x32_f16(false, pf, false, bf1, (short)0, acc1, false, false);
        acc2 = __builtin_amdgcn_wmma_f32_16x16x32_f16(false, pf, false, bf2, (short)0, acc2, false, false);
        acc3 = __builtin_amdgcn_wmma_f32_16x16x32_f16(false, pf, false, bf3, (short)0, acc3, false, false);
    }

    // Epilogue: ELU + head-concat store.
    // C/D layout: VGPR r -> M = r + (l<16 ? 0 : 8), N = (l&15) + 16*nt
    const int gmBase = m0 + 16 * w + ((l < 16) ? 0 : 8);
    const int nBase  = (l & 15);
    v8f accs[4] = {acc0, acc1, acc2, acc3};
#pragma unroll
    for (int nt = 0; nt < 4; ++nt) {
#pragma unroll
        for (int r = 0; r < 8; ++r) {
            float v = accs[nt][r];
            v = v > 0.f ? v : (__expf(v) - 1.f);
            out[(size_t)(gmBase + r) * outStride + h * DOUT + nBase + nt * 16] = v;
        }
    }
}

// ---------------------------------------------------------------------------
// Host-side launch
// ---------------------------------------------------------------------------
extern "C" void kernel_launch(void* const* d_in, const int* in_sizes, int n_in,
                              void* d_out, int out_size, void* d_ws, size_t ws_size,
                              hipStream_t stream)
{
    const float* in1   = (const float*)d_in[0];   // [8192,128]
    const float* in2   = (const float*)d_in[1];   // [4096,128]
    const float* A     = (const float*)d_in[2];   // [4096,8192]
    const float* Ws    = (const float*)d_in[3];   // [4,128,64]
    const float* Wt    = (const float*)d_in[4];   // [4,128,64]
    const float* a_src = (const float*)d_in[5];   // [4,64]
    const float* a_tgt = (const float*)d_in[6];   // [4,64]
    float* out = (float*)d_out;                   // out_st [8192,256] ++ out_ts [4096,256]

    char* w = (char*)d_ws;
    size_t off = 0;
    _Float16* hs_h = (_Float16*)(w + off); off += (size_t)NH * N_S * DOUT * 2; // 4 MiB
    _Float16* ht_h = (_Float16*)(w + off); off += (size_t)NH * N_T * DOUT * 2; // 2 MiB
    float* ss   = (float*)(w + off); off += (size_t)NH * N_S * 4;
    float* st   = (float*)(w + off); off += (size_t)NH * N_T * 4;
    float* m_ts = (float*)(w + off); off += (size_t)NH * N_T * 4;
    float* d_ts = (float*)(w + off); off += (size_t)NH * N_T * 4;
    float* m_st = (float*)(w + off); off += (size_t)NH * N_S * 4;
    float* d_st = (float*)(w + off); off += (size_t)NH * N_S * 4;

    // 1) projections + additive scores
    proj_kernel<<<dim3(N_S, NH, 2), 64, 0, stream>>>(in1, in2, Ws, Wt, a_src, a_tgt,
                                                     hs_h, ht_h, ss, st);
    // 2) softmax stats per target row (normalize over s)
    row_stats_kernel<<<N_T, 256, 0, stream>>>(A, st, ss, m_ts, d_ts);
    // 3) softmax stats per source column (normalize over t)
    col_stats_kernel<<<N_S / 256, 256, 0, stream>>>(A, st, ss, m_st, d_st);

    // 4) h_ts = softmax_s(e) @ hs  -> out_ts region (after out_st)
    attn_wmma_kernel<<<dim3(N_T / 64, NH), 128, 0, stream>>>(
        A, /*transA=*/0, st, ss, m_ts, d_ts, hs_h, N_T, N_S,
        out + (size_t)N_S * NH * DOUT, NH * DOUT);

    // 5) h_st = softmax_t(e)^T @ ht -> out_st region (start of d_out)
    attn_wmma_kernel<<<dim3(N_S / 64, NH), 128, 0, stream>>>(
        A, /*transA=*/1, ss, st, m_st, d_st, ht_h, N_S, N_T,
        out, NH * DOUT);
}
